// GraphEncoder_59313498358399
// MI455X (gfx1250) — matrix operand order repaired and checked by reading, not compiled, not run on hardware
//
#include <hip/hip_runtime.h>

typedef __bf16 bf16;
typedef __attribute__((ext_vector_type(16))) bf16 v16bf;
typedef __attribute__((ext_vector_type(8)))  bf16 v8bf;
typedef __attribute__((ext_vector_type(8)))  float v8f;

#define B_TOT 32768

// ---------- workspace layout (bf16 element offsets) ----------
#define OFF_W1T_BASE 0          // [256][320]
#define OFF_W1T_HIP  81920      // [256][96]
#define OFF_W1T_THI  106496
#define OFF_W1T_CAL  131072
#define OFF_W2T_BASE 155648     // [64][256]
#define OFF_W2T_HIP  172032
#define OFF_W2T_THI  188416
#define OFF_W2T_CAL  204800
#define OFF_WKT      221184     // [64][64]
#define OFF_WQT      225280
#define OFF_WVT      229376
#define OFF_WST      233472
#define OFF_WK2T     237568     // [32][64]
#define OFF_WQ2T     239616
#define OFF_WV2T     241664
#define OFF_WS2T     243712
#define OFF_NODES    245760     // [32768][13][64] bf16

// ---------- helpers ----------
__device__ __forceinline__ bf16 f2bf(float f) {
  unsigned u = __builtin_bit_cast(unsigned, f);
  u += 0x7FFFu + ((u >> 16) & 1u);               // round-to-nearest-even
  unsigned short h = (unsigned short)(u >> 16);
  return __builtin_bit_cast(bf16, h);
}

__device__ __forceinline__ v8f zero8() {
  v8f z = {0.f, 0.f, 0.f, 0.f, 0.f, 0.f, 0.f, 0.f};
  return z;
}

__device__ __forceinline__ v16bf cat88(v8bf lo, v8bf hi) {
  return __builtin_shufflevector(lo, hi, 0,1,2,3,4,5,6,7,8,9,10,11,12,13,14,15);
}

// A fragment: 16x32 bf16, row-major source with leading dim LDA.
// lane L(0..15): row L, K = {kb..kb+7, 16+kb..16+kb+7} with kb = (L>>4)*8
template<int LDA>
__device__ __forceinline__ v16bf load_a(const bf16* A) {
  int lane = threadIdx.x & 31;
  const bf16* p = A + (lane & 15) * LDA + ((lane >> 4) << 3);
  v8bf lo = *(const v8bf*)p;
  v8bf hi = *(const v8bf*)(p + 16);
  return cat88(lo, hi);
}

// A fragment from a [13][64] global tile, rows 13..15 clamped to row 12 (unused rows)
__device__ __forceinline__ v16bf load_a_clamp13(const bf16* A, int k0) {
  int lane = threadIdx.x & 31;
  int row = lane & 15; row = row < 13 ? row : 12;
  const bf16* p = A + row * 64 + k0 + ((lane >> 4) << 3);
  v8bf lo = *(const v8bf*)p;
  v8bf hi = *(const v8bf*)(p + 16);
  return cat88(lo, hi);
}

// B fragment: 32x16 bf16 from transposed weights BT[N][K] (leading dim LDB).
// lane L(0..15): col L, K = {kb..kb+7, 16+kb..16+kb+7} with kb = (L>>4)*8
template<int LDB>
__device__ __forceinline__ v16bf load_bT(const bf16* BT) {
  int lane = threadIdx.x & 31;
  const bf16* p = BT + (lane & 15) * LDB + ((lane >> 4) << 3);
  return cat88(*(const v8bf*)p, *(const v8bf*)(p + 16));
}

__device__ __forceinline__ v8f wmma_bf(v16bf a, v16bf b, v8f c) {
  return __builtin_amdgcn_wmma_f32_16x16x32_bf16(false, a, false, b, (short)0, c, false, false);
}

__device__ __forceinline__ float sigm(float x) { return 1.0f / (1.0f + __expf(-x)); }

// ---------- weight prep: fp32 [K][N] -> bf16 [N][Kpad] (transposed, zero-padded K) ----------
struct PrepDesc { const float* src; bf16* dst; int K; int N; int Kpad; };
struct PrepArgs { PrepDesc d[16]; };

__global__ void prep_kernel(PrepArgs args) {
  PrepDesc d = args.d[blockIdx.y];
  int total = d.N * d.Kpad;
  for (int i = blockIdx.x * blockDim.x + threadIdx.x; i < total;
       i += gridDim.x * blockDim.x) {
    int n = i / d.Kpad;
    int k = i - n * d.Kpad;
    float v = (k < d.K) ? d.src[k * d.N + n] : 0.f;
    d.dst[i] = f2bf(v);
  }
}

// ---------- base MLP: [B,320] -> relu -> [B,256] -> [B,64] (node 0) ----------
__global__ void __launch_bounds__(128)
mlp_base_kernel(const float* __restrict__ obs, const bf16* __restrict__ W1T,
                const float* __restrict__ b1, const bf16* __restrict__ W2T,
                const float* __restrict__ b2, bf16* __restrict__ nodes) {
  __shared__ bf16 Xs[64 * 160];   // half of K at a time
  __shared__ bf16 Hs[64 * 256];
  const int tid = threadIdx.x;
  const int wave = tid >> 5;
  const int lane = tid & 31;
  const int rowbase = blockIdx.x * 64;
  const int colc = lane & 15, hi8 = (lane >> 4) * 8;

  v8f acc[16];
  #pragma unroll
  for (int nt = 0; nt < 16; ++nt) acc[nt] = zero8();

  for (int kh = 0; kh < 2; ++kh) {
    // gather NODE_BASE permutation of obs into LDS (bf16)
    for (int i = tid; i < 64 * 160; i += 128) {
      int r = i / 160, cl = i - r * 160;
      int c = kh * 160 + cl;
      int h = c >> 4, j = c & 15;
      int col = (j < 12) ? j : (36 + j);            // NODE_BASE: 0..11,48..51
      Xs[i] = f2bf(obs[(size_t)(rowbase + r) * 1040 + h * 52 + col]);
    }
    __syncthreads();
    for (int k0 = 0; k0 < 160; k0 += 32) {
      v16bf a = load_a<160>(Xs + (wave * 16) * 160 + k0);
      #pragma unroll
      for (int nt = 0; nt < 16; ++nt) {
        v16bf b = load_bT<320>(W1T + (nt * 16) * 320 + kh * 160 + k0);
        acc[nt] = wmma_bf(a, b, acc[nt]);
      }
    }
    __syncthreads();
  }
  // bias + relu -> Hs bf16
  #pragma unroll
  for (int nt = 0; nt < 16; ++nt) {
    float bb = b1[nt * 16 + colc];
    #pragma unroll
    for (int r = 0; r < 8; ++r) {
      float v = acc[nt][r] + bb;
      Hs[(wave * 16 + r + hi8) * 256 + nt * 16 + colc] = f2bf(v > 0.f ? v : 0.f);
    }
  }
  // layer 2: K=256, N=64
  v8f a2[4];
  #pragma unroll
  for (int nt = 0; nt < 4; ++nt) a2[nt] = zero8();
  for (int k0 = 0; k0 < 256; k0 += 32) {
    v16bf a = load_a<256>(Hs + (wave * 16) * 256 + k0);
    #pragma unroll
    for (int nt = 0; nt < 4; ++nt) {
      v16bf b = load_bT<256>(W2T + (nt * 16) * 256 + k0);
      a2[nt] = wmma_bf(a, b, a2[nt]);
    }
  }
  #pragma unroll
  for (int nt = 0; nt < 4; ++nt) {
    float bb = b2[nt * 16 + colc];
    #pragma unroll
    for (int r = 0; r < 8; ++r) {
      int batch = rowbase + wave * 16 + r + hi8;
      nodes[(size_t)batch * 832 + nt * 16 + colc] = f2bf(a2[nt][r] + bb);
    }
  }
}

// ---------- leg MLP: [B*4,80(pad96)] -> relu -> 256 -> 64 (nodes 1+4w..4+4w) ----------
__global__ void __launch_bounds__(128)
mlp_leg_kernel(const float* __restrict__ obs, const bf16* __restrict__ W1T,
               const float* __restrict__ b1, const bf16* __restrict__ W2T,
               const float* __restrict__ b2, bf16* __restrict__ nodes, int which) {
  __shared__ bf16 Xs[64 * 96];
  __shared__ bf16 Hs[64 * 256];
  const int tid = threadIdx.x;
  const int wave = tid >> 5;
  const int lane = tid & 31;
  const int rowbase = blockIdx.x * 64;
  const int colc = lane & 15, hi8 = (lane >> 4) * 8;

  for (int i = tid; i < 64 * 96; i += 128) {
    int r = i / 96, c = i - r * 96;
    int gr = rowbase + r;
    int bidx = gr >> 2, n = gr & 3;
    float v = 0.f;
    if (c < 80) {
      int h = c >> 2, t = c & 3;
      int col = (t < 3) ? (12 + which + 12 * t + 3 * n) : (48 + n);
      v = obs[(size_t)bidx * 1040 + h * 52 + col];
    }
    Xs[i] = f2bf(v);
  }
  __syncthreads();

  v8f acc[16];
  #pragma unroll
  for (int nt = 0; nt < 16; ++nt) acc[nt] = zero8();
  for (int k0 = 0; k0 < 96; k0 += 32) {
    v16bf a = load_a<96>(Xs + (wave * 16) * 96 + k0);
    #pragma unroll
    for (int nt = 0; nt < 16; ++nt) {
      v16bf b = load_bT<96>(W1T + (nt * 16) * 96 + k0);
      acc[nt] = wmma_bf(a, b, acc[nt]);
    }
  }
  #pragma unroll
  for (int nt = 0; nt < 16; ++nt) {
    float bb = b1[nt * 16 + colc];
    #pragma unroll
    for (int r = 0; r < 8; ++r) {
      float v = acc[nt][r] + bb;
      Hs[(wave * 16 + r + hi8) * 256 + nt * 16 + colc] = f2bf(v > 0.f ? v : 0.f);
    }
  }
  v8f a2[4];
  #pragma unroll
  for (int nt = 0; nt < 4; ++nt) a2[nt] = zero8();
  for (int k0 = 0; k0 < 256; k0 += 32) {
    v16bf a = load_a<256>(Hs + (wave * 16) * 256 + k0);
    #pragma unroll
    for (int nt = 0; nt < 4; ++nt) {
      v16bf b = load_bT<256>(W2T + (nt * 16) * 256 + k0);
      a2[nt] = wmma_bf(a, b, a2[nt]);
    }
  }
  #pragma unroll
  for (int nt = 0; nt < 4; ++nt) {
    float bb = b2[nt * 16 + colc];
    #pragma unroll
    for (int r = 0; r < 8; ++r) {
      int gr = rowbase + wave * 16 + r + hi8;
      int batch = gr >> 2, n = gr & 3;
      int node = 1 + which * 4 + n;
      nodes[(size_t)batch * 832 + node * 64 + nt * 16 + colc] = f2bf(a2[nt][r] + bb);
    }
  }
}

// ---------- fused graph convs: one wave per batch graph ----------
__global__ void __launch_bounds__(128)
gconv_kernel(const bf16* __restrict__ nodes,
             const bf16* WkT, const bf16* WqT, const bf16* WvT, const bf16* WsT,
             const float* bk, const float* bq, const float* bv, const float* gb1,
             const bf16* Wk2T, const bf16* Wq2T, const bf16* Wv2T, const bf16* Ws2T,
             const float* bk2, const float* bq2, const float* bv2, const float* gb2,
             float* __restrict__ out) {
  __shared__ float fs[4 * 4 * 832];   // per wave: K,Q,V,S each [13][64] f32
  __shared__ bf16 x2s[4 * 1024];      // per wave: [16][64] bf16 (rows 13..15 zero)
  const int wave = threadIdx.x >> 5;
  const int lane = threadIdx.x & 31;
  const size_t batch = (size_t)blockIdx.x * 4 + wave;
  float* Ks = fs + wave * (4 * 832);
  float* Qs = Ks + 832;
  float* Vs = Qs + 832;
  float* Ss = Vs + 832;
  bf16* X2 = x2s + wave * 1024;
  const bf16* xb = nodes + batch * 832;
  const int colc = lane & 15, hi8 = (lane >> 4) * 8;

  // ---- stage 1: k,q,v,skip = x @ W (13x64 * 64x64), f32 into LDS ----
  auto gemm1 = [&](const bf16* WT, const float* bias, float* dst) {
    #pragma unroll
    for (int nt = 0; nt < 4; ++nt) {
      v8f c = zero8();
      #pragma unroll
      for (int k0 = 0; k0 < 64; k0 += 32) {
        v16bf a = load_a_clamp13(xb, k0);
        v16bf b = load_bT<64>(WT + (nt * 16) * 64 + k0);
        c = wmma_bf(a, b, c);
      }
      float bb = bias ? bias[nt * 16 + colc] : 0.f;
      #pragma unroll
      for (int r = 0; r < 8; ++r) {
        int rr = r + hi8;
        if (rr < 13) dst[rr * 64 + nt * 16 + colc] = c[r] + bb;
      }
    }
  };
  gemm1(WkT, bk, Ks);
  gemm1(WqT, bq, Qs);
  gemm1(WvT, bv, Vs);
  gemm1(WsT, nullptr, Ss);
  __syncthreads();

  // ---- gated aggregation 1 + relu -> X2 (bf16). Topology: node n<-0 (n>=1), n<-(n-4) (n>=5)
  for (int i = lane; i < 1024; i += 32) {
    int n = i >> 6, c = i & 63;
    float val = 0.f;
    if (n < 13) {
      float y = 0.f;
      if (n >= 1) {
        float kk = Ks[n * 64 + c];
        y = sigm(kk + Qs[c]) * Vs[c];
        if (n >= 5) {
          int s = n - 4;
          y += sigm(kk + Qs[s * 64 + c]) * Vs[s * 64 + c];
        }
      }
      val = y + Ss[n * 64 + c] + gb1[c];
      val = val > 0.f ? val : 0.f;
    }
    X2[i] = f2bf(val);
  }
  __syncthreads();

  // ---- stage 2: 13x64 * 64x32, reuse LDS float buffers ----
  auto gemm2 = [&](const bf16* WT, const float* bias, float* dst) {
    #pragma unroll
    for (int nt = 0; nt < 2; ++nt) {
      v8f c = zero8();
      #pragma unroll
      for (int k0 = 0; k0 < 64; k0 += 32) {
        v16bf a = load_a<64>(X2 + k0);
        v16bf b = load_bT<64>(WT + (nt * 16) * 64 + k0);
        c = wmma_bf(a, b, c);
      }
      float bb = bias ? bias[nt * 16 + colc] : 0.f;
      #pragma unroll
      for (int r = 0; r < 8; ++r) {
        int rr = r + hi8;
        if (rr < 13) dst[rr * 32 + nt * 16 + colc] = c[r] + bb;
      }
    }
  };
  gemm2(Wk2T, bk2, Ks);
  gemm2(Wq2T, bq2, Qs);
  gemm2(Wv2T, bv2, Vs);
  gemm2(Ws2T, nullptr, Ss);
  __syncthreads();

  // ---- gated aggregation 2 (no relu) -> out [B][13][32] f32 ----
  for (int i = lane; i < 416; i += 32) {
    int n = i >> 5, c = i & 31;
    float y = 0.f;
    if (n >= 1) {
      float kk = Ks[n * 32 + c];
      y = sigm(kk + Qs[c]) * Vs[c];
      if (n >= 5) {
        int s = n - 4;
        y += sigm(kk + Qs[s * 32 + c]) * Vs[s * 32 + c];
      }
    }
    out[batch * 416 + i] = y + Ss[n * 32 + c] + gb2[c];
  }
}

// ---------- host launcher ----------
extern "C" void kernel_launch(void* const* d_in, const int* in_sizes, int n_in,
                              void* d_out, int out_size, void* d_ws, size_t ws_size,
                              hipStream_t stream) {
  if (n_in < 33) return;
  const float* obs = (const float*)d_in[0];
  bf16* ws = (bf16*)d_ws;

  PrepArgs pa;
  pa.d[0]  = { (const float*)d_in[1],  ws + OFF_W1T_BASE, 320, 256, 320 };
  pa.d[1]  = { (const float*)d_in[5],  ws + OFF_W1T_HIP,   80, 256,  96 };
  pa.d[2]  = { (const float*)d_in[9],  ws + OFF_W1T_THI,   80, 256,  96 };
  pa.d[3]  = { (const float*)d_in[13], ws + OFF_W1T_CAL,   80, 256,  96 };
  pa.d[4]  = { (const float*)d_in[3],  ws + OFF_W2T_BASE, 256,  64, 256 };
  pa.d[5]  = { (const float*)d_in[7],  ws + OFF_W2T_HIP,  256,  64, 256 };
  pa.d[6]  = { (const float*)d_in[11], ws + OFF_W2T_THI,  256,  64, 256 };
  pa.d[7]  = { (const float*)d_in[15], ws + OFF_W2T_CAL,  256,  64, 256 };
  pa.d[8]  = { (const float*)d_in[17], ws + OFF_WKT,       64,  64,  64 };
  pa.d[9]  = { (const float*)d_in[19], ws + OFF_WQT,       64,  64,  64 };
  pa.d[10] = { (const float*)d_in[21], ws + OFF_WVT,       64,  64,  64 };
  pa.d[11] = { (const float*)d_in[23], ws + OFF_WST,       64,  64,  64 };
  pa.d[12] = { (const float*)d_in[25], ws + OFF_WK2T,      64,  32,  64 };
  pa.d[13] = { (const float*)d_in[27], ws + OFF_WQ2T,      64,  32,  64 };
  pa.d[14] = { (const float*)d_in[29], ws + OFF_WV2T,      64,  32,  64 };
  pa.d[15] = { (const float*)d_in[31], ws + OFF_WS2T,      64,  32,  64 };
  prep_kernel<<<dim3(320, 16), 256, 0, stream>>>(pa);

  bf16* nodes = ws + OFF_NODES;
  mlp_base_kernel<<<B_TOT / 64, 128, 0, stream>>>(
      obs, ws + OFF_W1T_BASE, (const float*)d_in[2],
      ws + OFF_W2T_BASE, (const float*)d_in[4], nodes);
  mlp_leg_kernel<<<(B_TOT * 4) / 64, 128, 0, stream>>>(
      obs, ws + OFF_W1T_HIP, (const float*)d_in[6],
      ws + OFF_W2T_HIP, (const float*)d_in[8], nodes, 0);
  mlp_leg_kernel<<<(B_TOT * 4) / 64, 128, 0, stream>>>(
      obs, ws + OFF_W1T_THI, (const float*)d_in[10],
      ws + OFF_W2T_THI, (const float*)d_in[12], nodes, 1);
  mlp_leg_kernel<<<(B_TOT * 4) / 64, 128, 0, stream>>>(
      obs, ws + OFF_W1T_CAL, (const float*)d_in[14],
      ws + OFF_W2T_CAL, (const float*)d_in[16], nodes, 2);

  gconv_kernel<<<B_TOT / 4, 128, 0, stream>>>(
      nodes,
      ws + OFF_WKT, ws + OFF_WQT, ws + OFF_WVT, ws + OFF_WST,
      (const float*)d_in[18], (const float*)d_in[20],
      (const float*)d_in[22], (const float*)d_in[24],
      ws + OFF_WK2T, ws + OFF_WQ2T, ws + OFF_WV2T, ws + OFF_WS2T,
      (const float*)d_in[26], (const float*)d_in[28],
      (const float*)d_in[30], (const float*)d_in[32],
      (float*)d_out);
}
